// Block_77515569758456
// MI455X (gfx1250) — matrix operand order repaired and checked by reading, not compiled, not run on hardware
//
#include <hip/hip_runtime.h>
#include <hip/hip_bf16.h>
#include <cmath>

// ---------------------------------------------------------------------------
// Problem constants (from reference)
// ---------------------------------------------------------------------------
#define NB   16
#define CIN  192
#define CO   256
#define CEMB 768
#define SS_  1024       // H*W
#define NH_  4          // heads
#define CPH_ 64

typedef __attribute__((ext_vector_type(16))) __bf16 v16bf;
typedef __attribute__((ext_vector_type(8)))  __bf16 v8bf;
typedef __attribute__((ext_vector_type(8)))  float  v8f;

static __device__ inline v8f wmma_bf16(v16bf a, v16bf b, v8f c) {
  return __builtin_amdgcn_wmma_f32_16x16x32_bf16(false, a, false, b, (short)0, c,
                                                 false, false);
}

// A fragment (16x32, row-major [m][k] tile, ld in elements, ld%8==0).
// Lane L: m = L%16 ; K runs [kb, kb+8) and [16+kb, 16+kb+8), kb = (L/16)*8.
static __device__ inline v16bf load_fragA(const __bf16* base, int ld) {
  const int lane = threadIdx.x & 31;
  const int m  = lane & 15;
  const int kb = (lane >> 4) << 3;
  union { v16bf v; v8bf h[2]; } u;
  u.h[0] = *(const v8bf*)(base + m * ld + kb);
  u.h[1] = *(const v8bf*)(base + m * ld + kb + 16);
  return u.v;
}

// B fragment (32x16) from row-major [n][k] tile (i.e. B^T).
// Lane L: n = L%16 ; K runs [kb, kb+8) and [16+kb, 16+kb+8), kb = (L/16)*8.
static __device__ inline v16bf load_fragB(const __bf16* base, int ld) {
  const int lane = threadIdx.x & 31;
  const int n  = lane & 15;
  const int kb = (lane >> 4) << 3;
  union { v16bf v; v8bf h[2]; } u;
  u.h[0] = *(const v8bf*)(base + n * ld + kb);
  u.h[1] = *(const v8bf*)(base + n * ld + kb + 16);
  return u.v;
}

static __device__ inline float blk_sum(float v) {
  __shared__ float red[8];
  #pragma unroll
  for (int o = 16; o >= 1; o >>= 1) v += __shfl_xor(v, o, 32);
  const int lane = threadIdx.x & 31, wv = threadIdx.x >> 5;
  if (lane == 0) red[wv] = v;
  __syncthreads();
  float r = red[0];
  #pragma unroll
  for (int i = 1; i < 8; ++i) r += red[i];
  __syncthreads();
  return r;
}

// ---------------------------------------------------------------------------
// Kernel 0: magnitude-preserving weight normalization -> bf16 (+ tap reorder)
//   scale = gain / (EPS*sqrt(fan_in) + ||w_row||)
// segments: 0 w_skip[256][192], 1 w_res0, 2 w_res1 (reorder (ci,kh,kw)->(tap,ci)),
//           3 w_qkv[768][256], 4 w_proj[256][256], 5 w_emb[256][768] (f32, transposed)
// ---------------------------------------------------------------------------
__global__ __launch_bounds__(256) void k_prep(
    const float* __restrict__ w_skip, const float* __restrict__ w_res0,
    const float* __restrict__ w_res1, const float* __restrict__ w_qkv,
    const float* __restrict__ w_proj, const float* __restrict__ w_emb,
    const float* __restrict__ emb_gain,
    __bf16* __restrict__ wskip_bf, __bf16* __restrict__ w0r,
    __bf16* __restrict__ w1r, __bf16* __restrict__ wqkv_bf,
    __bf16* __restrict__ wproj_bf, float* __restrict__ wembn_t) {
  const int b = blockIdx.x;
  int seg, row;
  if      (b < 256)  { seg = 0; row = b; }
  else if (b < 512)  { seg = 1; row = b - 256; }
  else if (b < 768)  { seg = 2; row = b - 512; }
  else if (b < 1536) { seg = 3; row = b - 768; }
  else if (b < 1792) { seg = 4; row = b - 1536; }
  else               { seg = 5; row = b - 1792; }
  const float* src; int len; float gain = 1.0f;
  switch (seg) {
    case 0:  src = w_skip + (size_t)row * 192;  len = 192;  break;
    case 1:  src = w_res0 + (size_t)row * 2304; len = 2304; break;
    case 2:  src = w_res1 + (size_t)row * 2304; len = 2304; break;
    case 3:  src = w_qkv  + (size_t)row * 256;  len = 256;  break;
    case 4:  src = w_proj + (size_t)row * 256;  len = 256;  break;
    default: src = w_emb  + (size_t)row * 768;  len = 768;  gain = emb_gain[0]; break;
  }
  float ss = 0.0f;
  for (int j = threadIdx.x; j < len; j += 256) { float v = src[j]; ss += v * v; }
  ss = blk_sum(ss);
  const float scale = gain / (1e-4f * sqrtf((float)len) + sqrtf(ss));
  for (int j = threadIdx.x; j < len; j += 256) {
    float v;
    if (seg == 1 || seg == 2) { int tap = j >> 8, ci = j & 255; v = src[ci * 9 + tap] * scale; }
    else v = src[j] * scale;
    switch (seg) {
      case 0:  wskip_bf[(size_t)row * 192 + j] = (__bf16)v; break;
      case 1:  w0r[(size_t)row * 2304 + j]     = (__bf16)v; break;
      case 2:  w1r[(size_t)row * 2304 + j]     = (__bf16)v; break;
      case 3:  wqkv_bf[(size_t)row * 256 + j]  = (__bf16)v; break;
      case 4:  wproj_bf[(size_t)row * 256 + j] = (__bf16)v; break;
      default: wembn_t[(size_t)j * 256 + row]  = v; break;
    }
  }
}

// c[n][co] = emb[n] . wembn[co] + 1
__global__ __launch_bounds__(256) void k_cvec(const float* __restrict__ emb,
                                              const float* __restrict__ wembn_t,
                                              float* __restrict__ cvec) {
  const int n = blockIdx.x, co = threadIdx.x;
  float s = 0.0f;
  for (int k = 0; k < CEMB; ++k) s += emb[n * CEMB + k] * wembn_t[k * 256 + co];
  cvec[n * 256 + co] = s + 1.0f;
}

// ---------------------------------------------------------------------------
// Kernel: skip 1x1 conv  xu[n][256][S] = Wskip[256][192] * x[n][192][S]
// ---------------------------------------------------------------------------
__global__ __launch_bounds__(256) void k_skip(const __bf16* __restrict__ Wn,
                                              const float* __restrict__ x,
                                              float* __restrict__ xu) {
  __shared__ __bf16 lA[128 * 40];
  __shared__ __bf16 lB[128 * 40];
  const int t = threadIdx.x;
  const int m0 = blockIdx.x * 128, p0 = blockIdx.y * 128, n = blockIdx.z;
  const int wv = t >> 5, lane = t & 31, wr = wv >> 1, wc = wv & 1;
  const v8f zero = {0, 0, 0, 0, 0, 0, 0, 0};
  v8f acc[2][4];
  #pragma unroll
  for (int i = 0; i < 2; ++i)
    #pragma unroll
    for (int j = 0; j < 4; ++j) acc[i][j] = zero;
  const float* xb = x + (size_t)n * CIN * SS_;
  for (int k0 = 0; k0 < CIN; k0 += 32) {
    { // A stage: 128 rows x 32
      const int row = t >> 1, hfa = t & 1;
      const __bf16* s = Wn + (size_t)(m0 + row) * CIN + k0 + hfa * 16;
      v8bf* d = (v8bf*)(lA + row * 40 + hfa * 16);
      d[0] = ((const v8bf*)s)[0];
      d[1] = ((const v8bf*)s)[1];
    }
    { // B stage (transpose + f32->bf16): 32 k x 128 p
      const int ci = t >> 3, pb = (t & 7) * 16;
      const float* s = xb + (size_t)(k0 + ci) * SS_ + p0 + pb;
      #pragma unroll
      for (int j = 0; j < 16; ++j) lB[(pb + j) * 40 + ci] = (__bf16)s[j];
    }
    __syncthreads();
    v16bf af[2], bfg[4];
    #pragma unroll
    for (int i = 0; i < 2; ++i) af[i] = load_fragA(lA + (wr * 32 + i * 16) * 40, 40);
    #pragma unroll
    for (int j = 0; j < 4; ++j) bfg[j] = load_fragB(lB + (wc * 64 + j * 16) * 40, 40);
    #pragma unroll
    for (int i = 0; i < 2; ++i)
      #pragma unroll
      for (int j = 0; j < 4; ++j) acc[i][j] = wmma_bf16(af[i], bfg[j], acc[i][j]);
    __syncthreads();
  }
  const int col = lane & 15, rh = (lane >> 4) * 8;
  float* ob = xu + (size_t)n * CO * SS_;
  #pragma unroll
  for (int i = 0; i < 2; ++i)
    #pragma unroll
    for (int j = 0; j < 4; ++j)
      #pragma unroll
      for (int r = 0; r < 8; ++r)
        ob[(size_t)(m0 + wr * 32 + i * 16 + rh + r) * SS_ + p0 + wc * 64 + j * 16 + col] =
            acc[i][j][r];
}

// Pixel norm over channels + mp_silu -> bf16
__global__ __launch_bounds__(256) void k_pixnorm(const float* __restrict__ xu,
                                                 float* __restrict__ xn,
                                                 __bf16* __restrict__ xs) {
  const int idx = blockIdx.x * 256 + threadIdx.x;   // n*S + p
  const int n = idx >> 10, p = idx & 1023;
  const float* col = xu + (size_t)n * CO * SS_ + p;
  float ss = 0.0f;
  for (int c = 0; c < CO; ++c) { float v = col[(size_t)c * SS_]; ss += v * v; }
  const float sc = 1.0f / (1e-4f + sqrtf(ss) * 0.0625f);   // alpha = 1/16
  float* on = xn + (size_t)n * CO * SS_ + p;
  __bf16* os = xs + (size_t)n * CO * SS_ + p;
  for (int c = 0; c < CO; ++c) {
    float v = col[(size_t)c * SS_] * sc;
    on[(size_t)c * SS_] = v;
    float sl = v / (1.0f + __expf(-v)) * 1.6778523f;        // mp_silu
    os[(size_t)c * SS_] = (__bf16)sl;
  }
}

// ---------------------------------------------------------------------------
// Implicit-GEMM 3x3 conv, K reordered as (tap, ci): K = 9*256, tile 32.
// MODE 0: out = bf16(mp_silu(acc * c[n][co]))            (aux = cvec)
// MODE 1: xa = mp_sum(xn, acc, 0.3) -> f32 + bf16        (aux = xn)
// ---------------------------------------------------------------------------
template <int MODE>
__global__ __launch_bounds__(256) void k_conv3(const __bf16* __restrict__ Wr,
                                               const __bf16* __restrict__ src,
                                               const float* __restrict__ aux,
                                               float* __restrict__ outf,
                                               __bf16* __restrict__ outb) {
  __shared__ __bf16 lA[128 * 40];
  __shared__ __bf16 lB[128 * 40];
  const int t = threadIdx.x;
  const int m0 = blockIdx.x * 128, p0 = blockIdx.y * 128, n = blockIdx.z;
  const int wv = t >> 5, lane = t & 31, wr = wv >> 1, wc = wv & 1;
  const v8f zero = {0, 0, 0, 0, 0, 0, 0, 0};
  v8f acc[2][4];
  #pragma unroll
  for (int i = 0; i < 2; ++i)
    #pragma unroll
    for (int j = 0; j < 4; ++j) acc[i][j] = zero;
  const __bf16 BZ = (__bf16)0.0f;
  const int arow = t >> 1, ahalf = t & 1;
  const int bci = t >> 3;            // 0..31
  const int bpb = (t & 7) * 16;      // 0..112
  const int bh = (p0 + bpb) >> 5;    // image row
  const int bw0 = bpb & 31;          // 0 or 16
  for (int kt = 0; kt < 72; ++kt) {
    const int tap = kt >> 3;
    const int kh = tap / 3, kw = tap % 3;
    const int ci0 = (kt & 7) * 32;
    { // A stage
      const __bf16* s = Wr + (size_t)(m0 + arow) * 2304 + kt * 32 + ahalf * 16;
      v8bf* d = (v8bf*)(lA + arow * 40 + ahalf * 16);
      d[0] = ((const v8bf*)s)[0];
      d[1] = ((const v8bf*)s)[1];
      if (kt + 1 < 72)
        __builtin_prefetch(Wr + (size_t)(m0 + arow) * 2304 + (kt + 1) * 32, 0, 1);
    }
    { // B stage with tap shift + zero padding
      const int hs = bh + kh - 1;
      const bool hok = (hs >= 0) && (hs < 32);
      const __bf16* s = src + ((size_t)n * CO + ci0 + bci) * SS_ + hs * 32 + (kw - 1);
      #pragma unroll
      for (int j = 0; j < 16; ++j) {
        const int w = bw0 + j + kw - 1;
        lB[(bpb + j) * 40 + bci] = (hok && w >= 0 && w < 32) ? s[bw0 + j] : BZ;
      }
    }
    __syncthreads();
    v16bf af[2], bfg[4];
    #pragma unroll
    for (int i = 0; i < 2; ++i) af[i] = load_fragA(lA + (wr * 32 + i * 16) * 40, 40);
    #pragma unroll
    for (int j = 0; j < 4; ++j) bfg[j] = load_fragB(lB + (wc * 64 + j * 16) * 40, 40);
    #pragma unroll
    for (int i = 0; i < 2; ++i)
      #pragma unroll
      for (int j = 0; j < 4; ++j) acc[i][j] = wmma_bf16(af[i], bfg[j], acc[i][j]);
    __syncthreads();
  }
  const int col = lane & 15, rh = (lane >> 4) * 8;
  #pragma unroll
  for (int i = 0; i < 2; ++i)
    #pragma unroll
    for (int j = 0; j < 4; ++j)
      #pragma unroll
      for (int r = 0; r < 8; ++r) {
        const int co = m0 + wr * 32 + i * 16 + rh + r;
        const int p = p0 + wc * 64 + j * 16 + col;
        const size_t idx = ((size_t)n * CO + co) * SS_ + p;
        float v = acc[i][j][r];
        if (MODE == 0) {
          v *= aux[n * 256 + co];
          outb[idx] = (__bf16)(v / (1.0f + __expf(-v)) * 1.6778523f);
        } else {
          const float r2 = (0.7f * aux[idx] + 0.3f * v) * 1.3130643f;  // mp_sum t=0.3
          outf[idx] = r2;
          outb[idx] = (__bf16)r2;
        }
      }
}

// ---------------------------------------------------------------------------
// qkv 1x1 conv: qkv_u[n][768][S] = Wqkv[768][256] * xab[n][256][S]
// ---------------------------------------------------------------------------
__global__ __launch_bounds__(256) void k_qkv(const __bf16* __restrict__ Wn,
                                             const __bf16* __restrict__ xab,
                                             float* __restrict__ qkv_u) {
  __shared__ __bf16 lA[128 * 40];
  __shared__ __bf16 lB[128 * 40];
  const int t = threadIdx.x;
  const int m0 = blockIdx.x * 128, p0 = blockIdx.y * 128, n = blockIdx.z;
  const int wv = t >> 5, lane = t & 31, wr = wv >> 1, wc = wv & 1;
  const v8f zero = {0, 0, 0, 0, 0, 0, 0, 0};
  v8f acc[2][4];
  #pragma unroll
  for (int i = 0; i < 2; ++i)
    #pragma unroll
    for (int j = 0; j < 4; ++j) acc[i][j] = zero;
  for (int k0 = 0; k0 < 256; k0 += 32) {
    { const int row = t >> 1, hfa = t & 1;
      const __bf16* s = Wn + (size_t)(m0 + row) * 256 + k0 + hfa * 16;
      v8bf* d = (v8bf*)(lA + row * 40 + hfa * 16);
      d[0] = ((const v8bf*)s)[0];
      d[1] = ((const v8bf*)s)[1]; }
    { const int ci = t >> 3, pb = (t & 7) * 16;
      const __bf16* s = xab + ((size_t)n * CO + k0 + ci) * SS_ + p0 + pb;
      #pragma unroll
      for (int j = 0; j < 16; ++j) lB[(pb + j) * 40 + ci] = s[j]; }
    __syncthreads();
    v16bf af[2], bfg[4];
    #pragma unroll
    for (int i = 0; i < 2; ++i) af[i] = load_fragA(lA + (wr * 32 + i * 16) * 40, 40);
    #pragma unroll
    for (int j = 0; j < 4; ++j) bfg[j] = load_fragB(lB + (wc * 64 + j * 16) * 40, 40);
    #pragma unroll
    for (int i = 0; i < 2; ++i)
      #pragma unroll
      for (int j = 0; j < 4; ++j) acc[i][j] = wmma_bf16(af[i], bfg[j], acc[i][j]);
    __syncthreads();
  }
  const int col = lane & 15, rh = (lane >> 4) * 8;
  float* ob = qkv_u + (size_t)n * 768 * SS_;
  #pragma unroll
  for (int i = 0; i < 2; ++i)
    #pragma unroll
    for (int j = 0; j < 4; ++j)
      #pragma unroll
      for (int r = 0; r < 8; ++r)
        ob[(size_t)(m0 + wr * 32 + i * 16 + rh + r) * SS_ + p0 + wc * 64 + j * 16 + col] =
            acc[i][j][r];
}

// Head-channel norm (over CPH=64, channel index = h*192 + c*3 + sel).
// Emits qT/kT as [n][h][s][64] and v as [n][h][64][s], bf16.
__global__ __launch_bounds__(256) void k_headnorm(const float* __restrict__ qkv,
                                                  __bf16* __restrict__ qT,
                                                  __bf16* __restrict__ kT,
                                                  __bf16* __restrict__ vB) {
  const int b = blockIdx.x;   // n*12 + h*3 + sel
  const int n = b / 12, rem = b % 12, h = rem / 3, sel = rem % 3;
  const size_t base = ((size_t)n * 768 + h * 192 + sel) * SS_;
  const int nh = n * NH_ + h;
  for (int s = threadIdx.x; s < SS_; s += 256) {
    float ss = 0.0f;
    for (int c = 0; c < CPH_; ++c) {
      const float v = qkv[base + (size_t)c * 3 * SS_ + s];
      ss += v * v;
    }
    const float scale = 1.0f / (1e-4f + sqrtf(ss) * 0.125f);   // alpha = 1/8
    for (int c = 0; c < CPH_; ++c) {
      const __bf16 bv = (__bf16)(qkv[base + (size_t)c * 3 * SS_ + s] * scale);
      if (sel == 0)      qT[((size_t)nh * SS_ + s) * 64 + c] = bv;
      else if (sel == 1) kT[((size_t)nh * SS_ + s) * 64 + c] = bv;
      else               vB[((size_t)nh * 64 + c) * SS_ + s] = bv;
    }
  }
}

// ---------------------------------------------------------------------------
// Flash attention per (n, head): BQ=128 queries/WG, BK=64 keys/step, d=64.
// Output o[n][s][h*64+c] bf16 ([pixel][channel] so proj B-stage is a copy).
// ---------------------------------------------------------------------------
__global__ __launch_bounds__(256) void k_attn(const __bf16* __restrict__ qT,
                                              const __bf16* __restrict__ kT,
                                              const __bf16* __restrict__ vB,
                                              __bf16* __restrict__ oB) {
  __shared__ __bf16 lQ[128 * 72];
  __shared__ __bf16 lK[64 * 72];
  __shared__ __bf16 lV[64 * 72];
  __shared__ __bf16 lP[128 * 72];
  const int t = threadIdx.x, lane = t & 31, wv = t >> 5;
  const int hf = lane >> 4, col = lane & 15;
  const int q0 = blockIdx.x * 128, h = blockIdx.y, n = blockIdx.z;
  const int nh = n * NH_ + h;
  { // stage Q once
    const int row = t >> 1, cb = (t & 1) * 32;
    const __bf16* s = qT + ((size_t)nh * SS_ + q0 + row) * 64 + cb;
    v8bf* d = (v8bf*)(lQ + row * 72 + cb);
    const v8bf* sv = (const v8bf*)s;
    d[0] = sv[0]; d[1] = sv[1]; d[2] = sv[2]; d[3] = sv[3];
  }
  const v8f zero = {0, 0, 0, 0, 0, 0, 0, 0};
  v8f O[4];
  float m[8], l[8];
  #pragma unroll
  for (int j = 0; j < 4; ++j) O[j] = zero;
  #pragma unroll
  for (int r = 0; r < 8; ++r) { m[r] = -INFINITY; l[r] = 0.0f; }

  for (int kb0 = 0; kb0 < SS_; kb0 += 64) {
    __syncthreads();
    { // stage K^T [k][c] and V [c][k]
      const int row = t >> 2, cb = (t & 3) * 16;
      const __bf16* sk = kT + ((size_t)nh * SS_ + kb0 + row) * 64 + cb;
      v8bf* dk = (v8bf*)(lK + row * 72 + cb);
      dk[0] = ((const v8bf*)sk)[0];
      dk[1] = ((const v8bf*)sk)[1];
      const __bf16* sv = vB + ((size_t)nh * 64 + row) * SS_ + kb0 + cb;
      v8bf* dv = (v8bf*)(lV + row * 72 + cb);
      dv[0] = ((const v8bf*)sv)[0];
      dv[1] = ((const v8bf*)sv)[1];
    }
    __syncthreads();
    // scores = Q . K^T  (this wave: 16 q rows x 64 k cols)
    v8f sc[4];
    #pragma unroll
    for (int j = 0; j < 4; ++j) sc[j] = zero;
    #pragma unroll
    for (int cs = 0; cs < 2; ++cs) {
      const v16bf a = load_fragA(lQ + (wv * 16) * 72 + cs * 32, 72);
      #pragma unroll
      for (int j = 0; j < 4; ++j) {
        const v16bf b = load_fragB(lK + (j * 16) * 72 + cs * 32, 72);
        sc[j] = wmma_bf16(a, b, sc[j]);
      }
    }
    // online softmax; q row of VGPR r is (wv*16 + r + 8*hf)
    #pragma unroll
    for (int r = 0; r < 8; ++r) {
      float mx = -INFINITY;
      #pragma unroll
      for (int j = 0; j < 4; ++j) mx = fmaxf(mx, sc[j][r]);
      mx *= 0.125f;                               // 1/sqrt(CPH)
      #pragma unroll
      for (int o = 8; o >= 1; o >>= 1) mx = fmaxf(mx, __shfl_xor(mx, o, 32));
      const float mnew = fmaxf(m[r], mx);
      const float corr = __expf(m[r] - mnew);
      float rsum = 0.0f;
      #pragma unroll
      for (int j = 0; j < 4; ++j) {
        const float p = __expf(sc[j][r] * 0.125f - mnew);
        lP[(wv * 16 + r + 8 * hf) * 72 + j * 16 + col] = (__bf16)p;
        rsum += p;
      }
      #pragma unroll
      for (int o = 8; o >= 1; o >>= 1) rsum += __shfl_xor(rsum, o, 32);
      l[r] = l[r] * corr + rsum;
      m[r] = mnew;
      #pragma unroll
      for (int j = 0; j < 4; ++j) O[j][r] *= corr;
    }
    // O += P . V   (lP region is wave-private; LDS ops are in-order per wave)
    #pragma unroll
    for (int ks = 0; ks < 2; ++ks) {
      const v16bf a = load_fragA(lP + (wv * 16) * 72 + ks * 32, 72);
      #pragma unroll
      for (int j = 0; j < 4; ++j) {
        const v16bf b = load_fragB(lV + (j * 16) * 72 + ks * 32, 72);
        O[j] = wmma_bf16(a, b, O[j]);
      }
    }
  }
  #pragma unroll
  for (int r = 0; r < 8; ++r) {
    const int q = q0 + wv * 16 + r + 8 * hf;
    const float inv = 1.0f / l[r];
    #pragma unroll
    for (int j = 0; j < 4; ++j)
      oB[((size_t)n * SS_ + q) * 256 + h * 64 + j * 16 + col] = (__bf16)(O[j][r] * inv);
  }
}

// ---------------------------------------------------------------------------
// proj 1x1 conv + mp_sum residual + clip -> d_out (f32)
// B source o[n][p][256] is already [pixel][k]: stage is a straight copy.
// ---------------------------------------------------------------------------
__global__ __launch_bounds__(256) void k_proj(const __bf16* __restrict__ Wn,
                                              const __bf16* __restrict__ oB,
                                              const float* __restrict__ xa,
                                              float* __restrict__ out) {
  __shared__ __bf16 lA[128 * 40];
  __shared__ __bf16 lB[128 * 40];
  const int t = threadIdx.x;
  const int m0 = blockIdx.x * 128, p0 = blockIdx.y * 128, n = blockIdx.z;
  const int wv = t >> 5, lane = t & 31, wr = wv >> 1, wc = wv & 1;
  const v8f zero = {0, 0, 0, 0, 0, 0, 0, 0};
  v8f acc[2][4];
  #pragma unroll
  for (int i = 0; i < 2; ++i)
    #pragma unroll
    for (int j = 0; j < 4; ++j) acc[i][j] = zero;
  for (int k0 = 0; k0 < 256; k0 += 32) {
    { const int row = t >> 1, hfa = t & 1;
      const __bf16* s = Wn + (size_t)(m0 + row) * 256 + k0 + hfa * 16;
      v8bf* d = (v8bf*)(lA + row * 40 + hfa * 16);
      d[0] = ((const v8bf*)s)[0];
      d[1] = ((const v8bf*)s)[1]; }
    { const int pr = t >> 1, kb = (t & 1) * 16;
      const __bf16* s = oB + ((size_t)n * SS_ + p0 + pr) * 256 + k0 + kb;
      v8bf* d = (v8bf*)(lB + pr * 40 + kb);
      d[0] = ((const v8bf*)s)[0];
      d[1] = ((const v8bf*)s)[1]; }
    __syncthreads();
    v16bf af[2], bfg[4];
    #pragma unroll
    for (int i = 0; i < 2; ++i) af[i] = load_fragA(lA + (wr * 32 + i * 16) * 40, 40);
    #pragma unroll
    for (int j = 0; j < 4; ++j) bfg[j] = load_fragB(lB + (wc * 64 + j * 16) * 40, 40);
    #pragma unroll
    for (int i = 0; i < 2; ++i)
      #pragma unroll
      for (int j = 0; j < 4; ++j) acc[i][j] = wmma_bf16(af[i], bfg[j], acc[i][j]);
    __syncthreads();
  }
  const int col = lane & 15, rh = (lane >> 4) * 8;
  #pragma unroll
  for (int i = 0; i < 2; ++i)
    #pragma unroll
    for (int j = 0; j < 4; ++j)
      #pragma unroll
      for (int r = 0; r < 8; ++r) {
        const int co = m0 + wr * 32 + i * 16 + rh + r;
        const int p = p0 + wc * 64 + j * 16 + col;
        const size_t idx = ((size_t)n * CO + co) * SS_ + p;
        float v = (0.7f * xa[idx] + 0.3f * acc[i][j][r]) * 1.3130643f;
        v = fminf(fmaxf(v, -256.0f), 256.0f);
        out[idx] = v;
      }
}

// ---------------------------------------------------------------------------
extern "C" void kernel_launch(void* const* d_in, const int* in_sizes, int n_in,
                              void* d_out, int out_size, void* d_ws, size_t ws_size,
                              hipStream_t stream) {
  const float* x        = (const float*)d_in[0];
  const float* emb      = (const float*)d_in[1];
  const float* w_skip   = (const float*)d_in[2];
  const float* w_res0   = (const float*)d_in[3];
  const float* w_res1   = (const float*)d_in[4];
  const float* w_emb    = (const float*)d_in[5];
  const float* emb_gain = (const float*)d_in[6];
  const float* w_qkv    = (const float*)d_in[7];
  const float* w_proj   = (const float*)d_in[8];
  float* out = (float*)d_out;

  size_t off = 0;
  auto alloc = [&](size_t bytes) -> void* {
    off = (off + 255) & ~(size_t)255;
    void* p = (char*)d_ws + off;
    off += bytes;
    return p;
  };
  const size_t ACT = (size_t)NB * CO * SS_;        // 4M elements
  __bf16* wskip_bf = (__bf16*)alloc((size_t)256 * 192 * 2);
  __bf16* w0r      = (__bf16*)alloc((size_t)256 * 2304 * 2);
  __bf16* w1r      = (__bf16*)alloc((size_t)256 * 2304 * 2);
  __bf16* wqkv_bf  = (__bf16*)alloc((size_t)768 * 256 * 2);
  __bf16* wproj_bf = (__bf16*)alloc((size_t)256 * 256 * 2);
  float*  wembn_t  = (float*) alloc((size_t)768 * 256 * 4);
  float*  cvec     = (float*) alloc((size_t)NB * 256 * 4);
  float*  xu       = (float*) alloc(ACT * 4);
  float*  xn       = (float*) alloc(ACT * 4);
  __bf16* xs       = (__bf16*)alloc(ACT * 2);
  __bf16* ys       = (__bf16*)alloc(ACT * 2);
  float*  xa       = (float*) alloc(ACT * 4);
  __bf16* xab      = (__bf16*)alloc(ACT * 2);
  float*  qkv_u    = (float*) alloc((size_t)NB * 768 * SS_ * 4);
  __bf16* qTb      = (__bf16*)alloc((size_t)NB * NH_ * SS_ * 64 * 2);
  __bf16* kTb      = (__bf16*)alloc((size_t)NB * NH_ * SS_ * 64 * 2);
  __bf16* vBb      = (__bf16*)alloc((size_t)NB * NH_ * 64 * SS_ * 2);
  __bf16* oBb      = (__bf16*)alloc((size_t)NB * SS_ * 256 * 2);

  k_prep<<<2048, 256, 0, stream>>>(w_skip, w_res0, w_res1, w_qkv, w_proj, w_emb,
                                   emb_gain, wskip_bf, w0r, w1r, wqkv_bf, wproj_bf,
                                   wembn_t);
  k_cvec<<<NB, 256, 0, stream>>>(emb, wembn_t, cvec);
  k_skip<<<dim3(2, 8, NB), 256, 0, stream>>>(wskip_bf, x, xu);
  k_pixnorm<<<64, 256, 0, stream>>>(xu, xn, xs);
  k_conv3<0><<<dim3(2, 8, NB), 256, 0, stream>>>(w0r, xs, cvec, xa /*unused*/, ys);
  k_conv3<1><<<dim3(2, 8, NB), 256, 0, stream>>>(w1r, ys, xn, xa, xab);
  k_qkv<<<dim3(6, 8, NB), 256, 0, stream>>>(wqkv_bf, xab, qkv_u);
  k_headnorm<<<NB * NH_ * 3, 256, 0, stream>>>(qkv_u, qTb, kTb, vBb);
  k_attn<<<dim3(8, NH_, NB), 256, 0, stream>>>(qTb, kTb, vBb, oBb);
  k_proj<<<dim3(2, 8, NB), 256, 0, stream>>>(wproj_bf, oBb, xa, out);
}
